// GraphConvBlock_26740466385633
// MI455X (gfx1250) — compile-verified, matrix operand order checked by reading, not run on hardware
//
#include <hip/hip_runtime.h>
#include <hip/hip_bf16.h>
#include <stdint.h>

// ---------------- problem constants ----------------
#define NNODES 8192
#define NEDGES 262144
#define DF     128      // d_model
#define DH     256      // d_inner
#define NEXP   64
#define MOE_MAX_TILES 1088   // 64 + 16384/16

typedef __attribute__((ext_vector_type(16))) __bf16 bf16x16;
typedef __attribute__((ext_vector_type(8)))  __bf16 bf16x8;
typedef __attribute__((ext_vector_type(8)))  float  f32x8;

// ---------------- WMMA fragment helpers ----------------
// A fragment (16x32 bf16, row-major source, lda in elements):
// lane&15 = row; lane>=16 -> kbase += 8; elements = K{kb..kb+7, kb+16..kb+23}
__device__ __forceinline__ bf16x16 load_a_frag(const __bf16* A, int lda, int row, int kbase) {
    const bf16x8* p0 = (const bf16x8*)(A + (size_t)row * lda + kbase);
    const bf16x8* p1 = (const bf16x8*)(A + (size_t)row * lda + kbase + 16);
    bf16x8 lo = *p0, hi = *p1;
    bf16x16 r;
#pragma unroll
    for (int i = 0; i < 8; ++i) { r[i] = lo[i]; r[8 + i] = hi[i]; }
    return r;
}

// Packed-B layout: fragment (nt, kt) of B[K][N]; per lane 16 contiguous bf16.
// index = ((nt*KT + kt)*32 + lane)*16 ; KT = K/32
__device__ __forceinline__ bf16x16 load_b_packed(const __bf16* Bp, int KT, int nt, int kt, int lane) {
    const bf16x16* p = (const bf16x16*)(Bp + (((size_t)(nt * KT + kt)) * 32 + lane) * 16);
    return *p;
}

__device__ __forceinline__ f32x8 wmma_bf16(bf16x16 a, bf16x16 b, f32x8 c) {
    return __builtin_amdgcn_wmma_f32_16x16x32_bf16(false, a, false, b, (short)0, c, false, false);
}

// ---------------- fp32 -> bf16 convert (A-side, row-major) ----------------
__global__ void cvt_bf16_kernel(const float* __restrict__ in, __bf16* __restrict__ out, int n) {
    int i = blockIdx.x * blockDim.x + threadIdx.x;
    if (i < n) out[i] = (__bf16)in[i];
}

// ---------------- fp32 -> packed bf16 B fragments (batched) ----------------
// one thread per (batch, fragment, lane); writes 16 contiguous bf16 (32 B)
__global__ void pack_b_kernel(const float* __restrict__ in, __bf16* __restrict__ out,
                              int K, int N, int batch) {
    int KT = K / 32, NT = N / 16;
    int frag_per_mat = NT * KT;
    int total = batch * frag_per_mat * 32;
    int t = blockIdx.x * blockDim.x + threadIdx.x;
    if (t >= total) return;
    int lane = t & 31;
    int f    = (t >> 5) % frag_per_mat;
    int b    = (t >> 5) / frag_per_mat;
    int kt   = f % KT;
    int nt   = f / KT;
    const float* Bm = in + (size_t)b * K * N;
    __bf16* op = out + ((size_t)b * frag_per_mat + f) * 32 * 16 + (size_t)lane * 16;
    int col = nt * 16 + (lane & 15);
    int kb  = kt * 32 + ((lane >> 4) << 3);
#pragma unroll
    for (int i = 0; i < 8; ++i) {
        op[i]     = (__bf16)Bm[(size_t)(kb + i) * N + col];
        op[8 + i] = (__bf16)Bm[(size_t)(kb + 16 + i) * N + col];
    }
}

// ---------------- generic bf16 WMMA GEMM: C[M,N] = A[M,K]@Bp + bias0 + bias1 -------
// blockDim = 128 (4 waves); grid = (M/16, ceil(N/64)); Bp is fragment-packed
__global__ void gemm_bf16_kernel(const __bf16* __restrict__ A, const __bf16* __restrict__ Bp,
                                 float* __restrict__ C, int M, int N, int K,
                                 const float* __restrict__ bias0, const float* __restrict__ bias1) {
    int wave = threadIdx.x >> 5, lane = threadIdx.x & 31;
    int m0 = blockIdx.x * 16;
    int n0 = (blockIdx.y * 4 + wave) * 16;
    if (n0 >= N) return;
    int KT    = K / 32;
    int half8 = (lane >> 4) << 3;   // 0 or 8
    int row   = m0 + (lane & 15);
    int col   = n0 + (lane & 15);
    f32x8 acc = {};
    for (int kt = 0; kt < KT; ++kt) {
        bf16x16 a = load_a_frag(A, K, row, kt * 32 + half8);
        bf16x16 b = load_b_packed(Bp, KT, n0 >> 4, kt, lane);
        acc = wmma_bf16(a, b, acc);
    }
    float badd = 0.f;
    if (bias0) badd += bias0[col];
    if (bias1) badd += bias1[col];
    int rbase = m0 + half8;
#pragma unroll
    for (int j = 0; j < 8; ++j) C[(size_t)(rbase + j) * N + col] = acc[j] + badd;
}

// ---------------- SPMM: agg[dst] += ew * support[src] (32 lanes per edge) ----------
__global__ void spmm_kernel(const int* __restrict__ src, const int* __restrict__ dst,
                            const float* __restrict__ ew, const float* __restrict__ support,
                            float* __restrict__ agg, int E) {
    int t = blockIdx.x * blockDim.x + threadIdx.x;
    int e = t >> 5;
    if (e >= E) return;
    int l = t & 31;
    float w = ew[e];
    const float4* sp = (const float4*)(support + (size_t)src[e] * DF);
    float4 v = sp[l];
    float* ag = agg + (size_t)dst[e] * DF + l * 4;
    atomicAdd(ag + 0, w * v.x);
    atomicAdd(ag + 1, w * v.y);
    atomicAdd(ag + 2, w * v.z);
    atomicAdd(ag + 3, w * v.w);
}

// ---------------- BatchNorm stats: per-feature scale/shift (128 blocks) ------------
__global__ void bn_stats_kernel(const float* __restrict__ x, const float* __restrict__ gamma,
                                const float* __restrict__ beta, float* __restrict__ scale,
                                float* __restrict__ shift, int Nrows) {
    int f = blockIdx.x;
    float s = 0.f, s2 = 0.f;
    for (int r = threadIdx.x; r < Nrows; r += blockDim.x) {
        float v = x[(size_t)r * DF + f];
        s += v; s2 += v * v;
    }
    __shared__ float sh[256], sh2[256];
    sh[threadIdx.x] = s; sh2[threadIdx.x] = s2;
    __syncthreads();
    for (int o = blockDim.x >> 1; o > 0; o >>= 1) {
        if (threadIdx.x < o) { sh[threadIdx.x] += sh[threadIdx.x + o]; sh2[threadIdx.x] += sh2[threadIdx.x + o]; }
        __syncthreads();
    }
    if (threadIdx.x == 0) {
        float mu  = sh[0] / Nrows;
        float var = sh2[0] / Nrows - mu * mu;
        float sc  = gamma[f] * rsqrtf(var + 1e-5f);
        scale[f] = sc;
        shift[f] = beta[f] - mu * sc;
    }
}

__global__ void bn_apply_kernel(const float* __restrict__ in, const float* __restrict__ scale,
                                const float* __restrict__ shift, float* __restrict__ out,
                                __bf16* __restrict__ out_bf, int n) {
    int i = blockIdx.x * blockDim.x + threadIdx.x;
    if (i >= n) return;
    int f = i & (DF - 1);
    float v = in[i] * scale[f] + shift[f];
    if (out)    out[i]    = v;
    if (out_bf) out_bf[i] = (__bf16)v;
}

// ---------------- gate top-2 + softmax + expert histogram --------------------------
__global__ void topk_kernel(const float* __restrict__ logits, const float* __restrict__ gate_b,
                            int* __restrict__ tki, float* __restrict__ tkp,
                            int* __restrict__ counts, int N) {
    int t = blockIdx.x * blockDim.x + threadIdx.x;
    if (t >= N) return;
    const float* l = logits + (size_t)t * NEXP;
    float b0 = -1e30f, b1 = -1e30f; int i0 = 0, i1 = 0;
    for (int e = 0; e < NEXP; ++e) {
        float v = l[e] + gate_b[e];
        if (v > b0)      { b1 = b0; i1 = i0; b0 = v; i0 = e; }
        else if (v > b1) { b1 = v; i1 = e; }
    }
    float e1 = __expf(b1 - b0);
    float inv = 1.f / (1.f + e1);
    tki[t * 2] = i0; tki[t * 2 + 1] = i1;
    tkp[t * 2] = inv; tkp[t * 2 + 1] = e1 * inv;
    atomicAdd(&counts[i0], 1);
    atomicAdd(&counts[i1], 1);
}

// ---------------- prefix-sum + flat tile work-map (single thread, 64 experts) ------
__global__ void scan_kernel(const int* __restrict__ counts, int* __restrict__ offsets,
                            int* __restrict__ cursors, int* __restrict__ tile_map,
                            int* __restrict__ num_tiles) {
    if (threadIdx.x != 0 || blockIdx.x != 0) return;
    int off = 0, nt = 0;
    for (int e = 0; e < NEXP; ++e) {
        offsets[e] = off;
        cursors[e] = 0;
        int c = counts[e];
        int tiles = (c + 15) >> 4;
        for (int ti = 0; ti < tiles; ++ti) tile_map[nt++] = (e << 16) | ti;
        off += c;
    }
    *num_tiles = nt;
}

__global__ void scatter_kernel(const int* __restrict__ tki, const float* __restrict__ tkp,
                               const int* __restrict__ offsets, int* __restrict__ cursors,
                               int* __restrict__ perm_tok, float* __restrict__ perm_w, int N) {
    int t = blockIdx.x * blockDim.x + threadIdx.x;
    if (t >= N) return;
#pragma unroll
    for (int k = 0; k < 2; ++k) {
        int e = tki[t * 2 + k];
        int pos = offsets[e] + atomicAdd(&cursors[e], 1);
        perm_tok[pos] = t;
        perm_w[pos]   = tkp[t * 2 + k];
    }
}

// ---------------- MoE expert FFN tile: 16 tokens, d->2d GELU ->d, scaled combine ---
// blockDim = 256 (8 waves); grid = MOE_MAX_TILES (over-launched, guarded)
__global__ void moe_kernel(const __bf16* __restrict__ x_bf, const __bf16* __restrict__ W1p,
                           const float* __restrict__ b1, const __bf16* __restrict__ W2p,
                           const float* __restrict__ b2, const int* __restrict__ perm_tok,
                           const float* __restrict__ perm_w, const int* __restrict__ offsets,
                           const int* __restrict__ counts, const int* __restrict__ tile_map,
                           const int* __restrict__ num_tiles, float* __restrict__ core) {
    __shared__ __bf16 xa[16 * DF];     // 4 KB
    __shared__ __bf16 hh[16 * DH];     // 8 KB
    __shared__ int    tok[16];
    __shared__ float  wrow[16];

    int bid = blockIdx.x;
    if (bid >= *num_tiles) return;
    int ent  = tile_map[bid];
    int e    = ent >> 16, tile = ent & 0xffff;
    int cnt  = counts[e];
    int base = offsets[e] + tile * 16;
    int rows = cnt - tile * 16; if (rows > 16) rows = 16;

    if (threadIdx.x < 16) {
        int r = threadIdx.x;
        if (r < rows) { tok[r] = perm_tok[base + r]; wrow[r] = perm_w[base + r]; }
        else          { tok[r] = -1;                 wrow[r] = 0.f; }
    }
    __syncthreads();

    // Stage 16 token rows (256 B each) into LDS via CDNA5 async memory->LDS path:
    // 256 threads x one 16 B chunk: row = tid>>4, chunk = tid&15.
    {
        int r = threadIdx.x >> 4;
        int c = threadIdx.x & 15;
        __bf16* ldst = xa + r * DF + c * 8;              // 8 bf16 = 16 B chunk
        if (tok[r] >= 0) {
            const __bf16* g = x_bf + (size_t)tok[r] * DF + c * 8;
            unsigned lds_off = (unsigned)(uintptr_t)ldst; // low 32 bits = LDS byte address
            asm volatile("global_load_async_to_lds_b128 %0, %1, off"
                         :: "v"(lds_off), "v"(g) : "memory");
        } else {
            uint4 z = {0u, 0u, 0u, 0u};
            *(uint4*)(void*)ldst = z;
        }
        asm volatile("s_wait_asynccnt 0x0" ::: "memory");
    }
    __syncthreads();

    int wave = threadIdx.x >> 5, lane = threadIdx.x & 31;
    int half8 = (lane >> 4) << 3;
    int rowl  = lane & 15;

    // phase 1: h = GELU(x @ W1e + b1e)  -> LDS (bf16). 16 N-tiles over 8 waves.
    const __bf16* W1e = W1p + (size_t)e * DF * DH;   // packed, KT = DF/32 = 4
    for (int nt = wave; nt < DH / 16; nt += 8) {
        int n0 = nt * 16;
        f32x8 acc = {};
#pragma unroll
        for (int kt = 0; kt < DF / 32; ++kt) {
            bf16x16 a = load_a_frag(xa, DF, rowl, kt * 32 + half8);
            bf16x16 b = load_b_packed(W1e, DF / 32, nt, kt, lane);
            acc = wmma_bf16(a, b, acc);
        }
        int col = n0 + rowl;
        float bias = b1[(size_t)e * DH + col];
        int rbase = half8;
#pragma unroll
        for (int j = 0; j < 8; ++j) {
            float v = acc[j] + bias;
            float g = 0.5f * v * (1.0f + erff(v * 0.70710678118654752f));  // exact GELU
            hh[(rbase + j) * DH + col] = (__bf16)g;
        }
    }
    __syncthreads();

    // phase 2: out = h @ W2e + b2e; core[tok] += cw * out. 8 N-tiles, 1 per wave.
    const __bf16* W2e = W2p + (size_t)e * DH * DF;   // packed, KT = DH/32 = 8
    {
        int n0 = wave * 16;
        f32x8 acc = {};
#pragma unroll
        for (int kt = 0; kt < DH / 32; ++kt) {
            bf16x16 a = load_a_frag(hh, DH, rowl, kt * 32 + half8);
            bf16x16 b = load_b_packed(W2e, DH / 32, wave, kt, lane);
            acc = wmma_bf16(a, b, acc);
        }
        int col = n0 + rowl;
        float bias = b2[(size_t)e * DF + col];
        int rbase = half8;
#pragma unroll
        for (int j = 0; j < 8; ++j) {
            int r = rbase + j;
            if (tok[r] >= 0)
                atomicAdd(&core[(size_t)tok[r] * DF + col], wrow[r] * (acc[j] + bias));
        }
    }
}

// =====================================================================
extern "C" void kernel_launch(void* const* d_in, const int* in_sizes, int n_in,
                              void* d_out, int out_size, void* d_ws, size_t ws_size,
                              hipStream_t stream) {
    const float* feats     = (const float*)d_in[0];
    const float* edge_w    = (const float*)d_in[1];
    const float* gc_weight = (const float*)d_in[2];
    const float* gc_bias   = (const float*)d_in[3];
    const float* res_W     = (const float*)d_in[4];
    const float* res_b     = (const float*)d_in[5];
    const float* bn1_g     = (const float*)d_in[6];
    const float* bn1_b     = (const float*)d_in[7];
    const float* gate_W    = (const float*)d_in[8];
    const float* gate_b    = (const float*)d_in[9];
    const float* W1        = (const float*)d_in[10];
    const float* b1        = (const float*)d_in[11];
    const float* W2        = (const float*)d_in[12];
    const float* b2        = (const float*)d_in[13];
    const float* bn2_g     = (const float*)d_in[14];
    const float* bn2_b     = (const float*)d_in[15];
    const int*   src       = (const int*)d_in[16];
    const int*   dst       = (const int*)d_in[17];
    float* out = (float*)d_out;

    // -------- workspace carving (256B aligned) --------
    char* ws = (char*)d_ws;
    size_t off = 0;
    auto alloc = [&](size_t bytes) -> char* {
        char* p = ws + off;
        off = (off + bytes + 255) & ~(size_t)255;
        return p;
    };
    __bf16* featsbf = (__bf16*)alloc((size_t)NNODES * DF * 2);
    __bf16* gcWp    = (__bf16*)alloc((size_t)DF * DF * 2);
    __bf16* resWp   = (__bf16*)alloc((size_t)DF * DF * 2);
    __bf16* gateWp  = (__bf16*)alloc((size_t)DF * NEXP * 2);
    __bf16* W1p     = (__bf16*)alloc((size_t)NEXP * DF * DH * 2);
    __bf16* W2p     = (__bf16*)alloc((size_t)NEXP * DH * DF * 2);
    float*  support = (float*)alloc((size_t)NNODES * DF * 4);
    float*  agg     = (float*)alloc((size_t)NNODES * DF * 4);   // new_feats
    float*  x       = (float*)alloc((size_t)NNODES * DF * 4);
    __bf16* xbf     = (__bf16*)alloc((size_t)NNODES * DF * 2);
    float*  logits  = (float*)alloc((size_t)NNODES * NEXP * 4);
    float*  scale1  = (float*)alloc(DF * 4);
    float*  shift1  = (float*)alloc(DF * 4);
    float*  scale2  = (float*)alloc(DF * 4);
    float*  shift2  = (float*)alloc(DF * 4);
    int*    tki     = (int*)alloc((size_t)NNODES * 2 * 4);
    float*  tkp     = (float*)alloc((size_t)NNODES * 2 * 4);
    int*    counts  = (int*)alloc(NEXP * 4);
    int*    offsets = (int*)alloc(NEXP * 4);
    int*    cursors = (int*)alloc(NEXP * 4);
    int*    tile_map  = (int*)alloc(MOE_MAX_TILES * 4);
    int*    num_tiles = (int*)alloc(4);
    int*    perm_tok  = (int*)alloc((size_t)NNODES * 2 * 4);
    float*  perm_w    = (float*)alloc((size_t)NNODES * 2 * 4);
    float*  core      = (float*)alloc((size_t)NNODES * DF * 4);
    (void)ws_size; (void)in_sizes; (void)n_in; (void)out_size;

    // 1) A-side bf16 conversion + B-side fragment packing
    cvt_bf16_kernel<<<(NNODES * DF + 255) / 256, 256, 0, stream>>>(feats, featsbf, NNODES * DF);
    auto pack = [&](const float* in, __bf16* o, int K, int N, int batch) {
        int total = batch * (N / 16) * (K / 32) * 32;
        pack_b_kernel<<<(total + 255) / 256, 256, 0, stream>>>(in, o, K, N, batch);
    };
    pack(gc_weight, gcWp,   DF, DF,   1);
    pack(res_W,     resWp,  DF, DF,   1);
    pack(gate_W,    gateWp, DF, NEXP, 1);
    pack(W1,        W1p,    DF, DH,   NEXP);
    pack(W2,        W2p,    DH, DF,   NEXP);

    // 2) support = feats @ gcW ; agg = feats @ resW + res_b + gc_bias
    dim3 gemm_grid(NNODES / 16, (DF + 63) / 64);
    gemm_bf16_kernel<<<gemm_grid, 128, 0, stream>>>(featsbf, gcWp, support,
                                                    NNODES, DF, DF, nullptr, nullptr);
    gemm_bf16_kernel<<<gemm_grid, 128, 0, stream>>>(featsbf, resWp, agg,
                                                    NNODES, DF, DF, res_b, gc_bias);

    // 3) SPMM scatter-add over edges
    spmm_kernel<<<(NEDGES * 32) / 256, 256, 0, stream>>>(src, dst, edge_w, support, agg, NEDGES);

    // 4) BN1 -> x (fp32 + bf16)
    bn_stats_kernel<<<DF, 256, 0, stream>>>(agg, bn1_g, bn1_b, scale1, shift1, NNODES);
    bn_apply_kernel<<<(NNODES * DF + 255) / 256, 256, 0, stream>>>(agg, scale1, shift1,
                                                                   x, xbf, NNODES * DF);

    // 5) gate logits = x @ gate_W (bias added in topk)
    dim3 gate_grid(NNODES / 16, (NEXP + 63) / 64);
    gemm_bf16_kernel<<<gate_grid, 128, 0, stream>>>(xbf, gateWp, logits,
                                                    NNODES, NEXP, DF, nullptr, nullptr);

    // 6) top-2 + softmax + bucketing
    hipMemsetAsync(counts, 0, NEXP * 4, stream);
    topk_kernel<<<(NNODES + 255) / 256, 256, 0, stream>>>(logits, gate_b, tki, tkp, counts, NNODES);
    scan_kernel<<<1, 1, 0, stream>>>(counts, offsets, cursors, tile_map, num_tiles);
    scatter_kernel<<<(NNODES + 255) / 256, 256, 0, stream>>>(tki, tkp, offsets, cursors,
                                                             perm_tok, perm_w, NNODES);

    // 7) core = x ; MoE experts accumulate into core
    hipMemcpyAsync(core, x, (size_t)NNODES * DF * 4, hipMemcpyDeviceToDevice, stream);
    moe_kernel<<<MOE_MAX_TILES, 256, 0, stream>>>(xbf, W1p, b1, W2p, b2, perm_tok, perm_w,
                                                  offsets, counts, tile_map, num_tiles, core);

    // 8) BN2 -> output
    bn_stats_kernel<<<DF, 256, 0, stream>>>(core, bn2_g, bn2_b, scale2, shift2, NNODES);
    bn_apply_kernel<<<(NNODES * DF + 255) / 256, 256, 0, stream>>>(core, scale2, shift2,
                                                                   out, nullptr, NNODES * DF);
}
